// GroupedQueryAttention_11536282157169
// MI455X (gfx1250) — matrix operand order repaired and checked by reading, not compiled, not run on hardware
//
#include <hip/hip_runtime.h>
#include <math.h>

// ---------------------------------------------------------------------------
// CDNA5 (gfx1250) wave32 WMMA types and fragment helpers
// ---------------------------------------------------------------------------
typedef __attribute__((ext_vector_type(4)))  float    v4f;
typedef __attribute__((ext_vector_type(8)))  float    v8f;
typedef __attribute__((ext_vector_type(4)))  _Float16 v4h;
typedef __attribute__((ext_vector_type(8)))  _Float16 v8h;
typedef __attribute__((ext_vector_type(16))) _Float16 v16h;

union Frag16 { v16h v; v8h h[2]; };

// A-matrix fragment (16x32 f16): lane r(<16) holds row r,
// K = {0..15}; lane r+16 holds row r,
// K = {16..31}.  Caller passes p = &A[row][k0 + half*16].
__device__ __forceinline__ v16h load_a_frag(const _Float16* p) {
  Frag16 f;
  f.h[0] = *(const v8h*)(p);
  f.h[1] = *(const v8h*)(p + 8);
  return f.v;
}

// D = A*B + C, f16 inputs, f32 accumulate (emits v_wmma_f32_16x16x32_f16)
#define WMMA_F16(a, b, c) \
  __builtin_amdgcn_wmma_f32_16x16x32_f16(false, (a), false, (b), (short)0, (c), false, false)

#define DIM_    2048
#define NHEADS_ 32
#define NKV_    8
#define HD_     64
#define HALF_   32

// ---------------------------------------------------------------------------
// fp32 -> f16 cast (vectorized)
// ---------------------------------------------------------------------------
__global__ __launch_bounds__(256) void cast_f32_f16(const float* __restrict__ in,
                                                    _Float16* __restrict__ out, int n4) {
  int i = blockIdx.x * 256 + threadIdx.x;
  if (i >= n4) return;
  v4f x = *(const v4f*)(in + (size_t)i * 4);
  v4h y;
  y[0] = (_Float16)x[0]; y[1] = (_Float16)x[1];
  y[2] = (_Float16)x[2]; y[3] = (_Float16)x[3];
  *(v4h*)(out + (size_t)i * 4) = y;
}

// ---------------------------------------------------------------------------
// C[M,N] = A[M,K] @ Bw[N,K]^T   (f16 in, f32 accumulate, f16 or f32 out)
// One wave per 32x32 tile, 4 WMMA per K-step of 32.
// ---------------------------------------------------------------------------
__global__ __launch_bounds__(256) void gemm_wmma_nt(const _Float16* __restrict__ A,
                                                    const _Float16* __restrict__ Bw,
                                                    void* __restrict__ Cout,
                                                    int M, int N, int K, int c_half) {
  const int lane = threadIdx.x & 31;
  const int wave = threadIdx.x >> 5;
  const int tilesN = N >> 5;
  const int tilesM = M >> 5;
  int w = blockIdx.x * 8 + wave;
  if (w >= tilesM * tilesN) return;
  const int m0 = (w / tilesN) << 5;
  const int n0 = (w % tilesN) << 5;
  const int half = lane >> 4, r = lane & 15;

  const _Float16* a0p = A + (size_t)(m0 + r) * K + half * 16;        // rows m0..m0+15
  const _Float16* a1p = A + (size_t)(m0 + 16 + r) * K + half * 16;   // rows m0+16..
  // B fragment: lane r -> column n0+r (= row of Bw); K contiguous 16 per half-wave
  const _Float16* b0p = Bw + (size_t)(n0 + r) * K + half * 16;
  const _Float16* b1p = Bw + (size_t)(n0 + 16 + r) * K + half * 16;

  v8f c00 = {}, c01 = {}, c10 = {}, c11 = {};
  for (int kk = 0; kk < K; kk += 32) {
    v16h a0 = load_a_frag(a0p + kk);
    v16h a1 = load_a_frag(a1p + kk);
    v16h b0 = *(const v16h*)(b0p + kk);   // 32B-aligned single load
    v16h b1 = *(const v16h*)(b1p + kk);
    __builtin_prefetch(a0p + kk + 256, 0, 3);   // global_prefetch_b8
    __builtin_prefetch(b0p + kk + 256, 0, 3);
    c00 = WMMA_F16(a0, b0, c00);
    c01 = WMMA_F16(a0, b1, c01);
    c10 = WMMA_F16(a1, b0, c10);
    c11 = WMMA_F16(a1, b1, c11);
  }

  // C layout: VGPR j -> row (j + 8*half), col = lane&15
  if (c_half) {
    _Float16* C = (_Float16*)Cout;
#pragma unroll
    for (int j = 0; j < 8; ++j) {
      size_t row = (size_t)(m0 + half * 8 + j);
      C[row * N + n0 + r]             = (_Float16)c00[j];
      C[row * N + n0 + 16 + r]        = (_Float16)c01[j];
      C[(row + 16) * N + n0 + r]      = (_Float16)c10[j];
      C[(row + 16) * N + n0 + 16 + r] = (_Float16)c11[j];
    }
  } else {
    float* C = (float*)Cout;
#pragma unroll
    for (int j = 0; j < 8; ++j) {
      size_t row = (size_t)(m0 + half * 8 + j);
      C[row * N + n0 + r]             = c00[j];
      C[row * N + n0 + 16 + r]        = c01[j];
      C[(row + 16) * N + n0 + r]      = c10[j];
      C[(row + 16) * N + n0 + 16 + r] = c11[j];
    }
  }
}

// ---------------------------------------------------------------------------
// RoPE: In [B*T, NH*64] (b,t,h,d) -> Out [B, NH, T, 64], optional scale
// ---------------------------------------------------------------------------
__global__ __launch_bounds__(256) void rope_kernel(const _Float16* __restrict__ In,
                                                   const float* __restrict__ Cos,
                                                   const float* __restrict__ Sin,
                                                   _Float16* __restrict__ Out,
                                                   int NH, float scale, int T, int total) {
  int idx = blockIdx.x * 256 + threadIdx.x;
  if (idx >= total) return;
  int p = idx & (HALF_ - 1);
  int tmp = idx >> 5;            // HALF_ == 32
  int hh = tmp % NH; tmp /= NH;
  int t = tmp % T;
  int b = tmp / T;
  const _Float16* src = In + ((size_t)(b * T + t) * NH + hh) * HD_ + 2 * p;
  float x0 = (float)src[0], x1 = (float)src[1];
  float c = Cos[(size_t)t * HALF_ + p];
  float s = Sin[(size_t)t * HALF_ + p];
  float o0 = (x0 * c - x1 * s) * scale;
  float o1 = (x0 * s + x1 * c) * scale;
  _Float16* dst = Out + (((size_t)(b * NH + hh) * T) + t) * HD_ + 2 * p;
  dst[0] = (_Float16)o0;
  dst[1] = (_Float16)o1;
}

// V [B*T, 8*64] (b,t,kv,d) -> Vt [B, 8, 64, T]  (transposed for B-fragments)
__global__ __launch_bounds__(256) void vprep_kernel(const _Float16* __restrict__ Vg,
                                                    _Float16* __restrict__ Vt,
                                                    int T, int total) {
  int idx = blockIdx.x * 256 + threadIdx.x;
  if (idx >= total) return;
  int d = idx & 63; int tmp = idx >> 6;
  int kv = tmp & 7; tmp >>= 3;
  int t = tmp % T;
  int b = tmp / T;
  Vt[(((size_t)(b * NKV_ + kv) * HD_) + d) * T + t] =
      Vg[((size_t)(b * T + t) * NKV_ + kv) * HD_ + d];
}

// ---------------------------------------------------------------------------
// Flash attention: one wave = 16 query rows of one (b, head).
// Qh [B,H,T,64] (pre-scaled by 1/8), Kh [B,KV,T,64], Vt [B,KV,64,T]
// Og [B*T, H*64] f16.  Online softmax, fp32 statistics and O accumulators.
// ---------------------------------------------------------------------------
__global__ __launch_bounds__(256) void attn_wmma(const _Float16* __restrict__ Qh,
                                                 const _Float16* __restrict__ Kh,
                                                 const _Float16* __restrict__ Vt,
                                                 _Float16* __restrict__ Og, int T) {
  __shared__ __align__(16) _Float16 Ps[8][16 * 32];   // per-wave P staging (8 KB)
  const int lane = threadIdx.x & 31;
  const int wave = threadIdx.x >> 5;
  const int half = lane >> 4, r = lane & 15;
  const int qtiles = T >> 4;
  int w = blockIdx.x * 8 + wave;
  if (w >= 2 * NHEADS_ * qtiles) return;
  const int qt = w % qtiles;
  const int bh = w / qtiles;          // b*32 + h
  const int h = bh & 31, b = bh >> 5;
  const int kv = h >> 2;              // N_REP = 4
  const int q0 = qt << 4;

  const _Float16* Qb = Qh + ((size_t)bh * T + q0) * HD_;
  const _Float16* Kb = Kh + ((size_t)(b * NKV_ + kv) * T) * HD_;
  const _Float16* Vb = Vt + ((size_t)(b * NKV_ + kv) * HD_) * T;

  // Q A-fragments (reused all iterations): dims 0..31 and 32..63
  const v16h qa0 = load_a_frag(Qb + (size_t)r * HD_ + half * 16);
  const v16h qa1 = load_a_frag(Qb + (size_t)r * HD_ + 32 + half * 16);

  float mrun[8], lrun[8];
  v8f oacc[4] = {{}, {}, {}, {}};
#pragma unroll
  for (int j = 0; j < 8; ++j) { mrun[j] = -3.0e38f; lrun[j] = 0.0f; }

  _Float16* P = &Ps[wave][0];

  for (int kb = 0; kb < q0 + 16; kb += 32) {
    // ---- S = Q @ K^T  (two 16-key tiles; all loads provably in-bounds) ----
    v8f z = {};
    v16h k0a = *(const v16h*)(Kb + (size_t)(kb + r) * HD_ + half * 16);
    v16h k0b = *(const v16h*)(Kb + (size_t)(kb + r) * HD_ + 32 + half * 16);
    v8f s0 = WMMA_F16(qa0, k0a, z);
    s0 = WMMA_F16(qa1, k0b, s0);
    v16h k1a = *(const v16h*)(Kb + (size_t)(kb + 16 + r) * HD_ + half * 16);
    v16h k1b = *(const v16h*)(Kb + (size_t)(kb + 16 + r) * HD_ + 32 + half * 16);
    v8f s1 = WMMA_F16(qa0, k1a, z);
    s1 = WMMA_F16(qa1, k1b, s1);

    // ---- causal mask + online softmax (rows live across 16-lane groups) ---
#pragma unroll
    for (int j = 0; j < 8; ++j) {
      const int qrow = q0 + half * 8 + j;
      if (kb + r > qrow)      s0[j] = -3.0e38f;
      if (kb + 16 + r > qrow) s1[j] = -3.0e38f;

      float t0 = fmaxf(s0[j], s1[j]);
      t0 = fmaxf(t0, __shfl_xor(t0, 1, 32));
      t0 = fmaxf(t0, __shfl_xor(t0, 2, 32));
      t0 = fmaxf(t0, __shfl_xor(t0, 4, 32));
      t0 = fmaxf(t0, __shfl_xor(t0, 8, 32));
      const float mnew = fmaxf(mrun[j], t0);
      const float alpha = __expf(mrun[j] - mnew);
      mrun[j] = mnew;
      const float p0 = __expf(s0[j] - mnew);
      const float p1 = __expf(s1[j] - mnew);
      float rs = p0 + p1;
      rs += __shfl_xor(rs, 1, 32);
      rs += __shfl_xor(rs, 2, 32);
      rs += __shfl_xor(rs, 4, 32);
      rs += __shfl_xor(rs, 8, 32);
      lrun[j] = lrun[j] * alpha + rs;
#pragma unroll
      for (int nt = 0; nt < 4; ++nt) oacc[nt][j] *= alpha;

      // stage P (C layout -> LDS, row-major 16x32)
      const int prow = half * 8 + j;
      P[prow * 32 + r]      = (_Float16)p0;
      P[prow * 32 + 16 + r] = (_Float16)p1;
    }

    // ---- read P back as A-fragment (wave-private LDS, in-order DS) --------
    Frag16 pf;
    pf.h[0] = *(const v8h*)&P[r * 32 + half * 16];
    pf.h[1] = *(const v8h*)&P[r * 32 + half * 16 + 8];

    // ---- O += P @ V  (Vt gives contiguous key runs per column) ------------
#pragma unroll
    for (int nt = 0; nt < 4; ++nt) {
      const _Float16* vp = Vb + (size_t)(nt * 16 + r) * T + kb + half * 16;
      v16h vf = *(const v16h*)vp;
      oacc[nt] = WMMA_F16(pf.v, vf, oacc[nt]);
    }
  }

  // ---- epilogue: normalize and scatter into [B*T, H*64] --------------------
#pragma unroll
  for (int j = 0; j < 8; ++j) {
    const float inv = 1.0f / lrun[j];
    const size_t row = (size_t)b * T + q0 + half * 8 + j;
    _Float16* op = Og + row * (size_t)DIM_ + (size_t)h * HD_;
    op[r]      = (_Float16)(oacc[0][j] * inv);
    op[16 + r] = (_Float16)(oacc[1][j] * inv);
    op[32 + r] = (_Float16)(oacc[2][j] * inv);
    op[48 + r] = (_Float16)(oacc[3][j] * inv);
  }
}

// ---------------------------------------------------------------------------
// Host launch
// ---------------------------------------------------------------------------
extern "C" void kernel_launch(void* const* d_in, const int* in_sizes, int n_in,
                              void* d_out, int out_size, void* d_ws, size_t ws_size,
                              hipStream_t stream) {
  (void)n_in; (void)out_size; (void)ws_size;
  const float* x  = (const float*)d_in[0];
  // d_in[1] = mask (unused: causality computed analytically)
  const float* wq = (const float*)d_in[2];
  const float* wk = (const float*)d_in[3];
  const float* wv = (const float*)d_in[4];
  const float* wo = (const float*)d_in[5];
  const float* cs = (const float*)d_in[6];
  const float* sn = (const float*)d_in[7];

  const int B = 2, DIM = DIM_, NH = NHEADS_, NKV = NKV_;
  const int KVD = NKV * HD_;                 // 512
  const int T = in_sizes[0] / (B * DIM);     // 2048
  const int BT = B * T;

  // workspace carve-out (256B aligned), total ~100 MB
  char* wp = (char*)d_ws;
  auto take = [&](size_t bytes) {
    char* p = wp;
    wp += (bytes + 255) & ~(size_t)255;
    return p;
  };
  _Float16* xh  = (_Float16*)take((size_t)BT * DIM * 2);
  _Float16* wqh = (_Float16*)take((size_t)DIM * DIM * 2);
  _Float16* wkh = (_Float16*)take((size_t)KVD * DIM * 2);
  _Float16* wvh = (_Float16*)take((size_t)KVD * DIM * 2);
  _Float16* woh = (_Float16*)take((size_t)DIM * DIM * 2);
  _Float16* Qg  = (_Float16*)take((size_t)BT * DIM * 2);
  _Float16* Kg  = (_Float16*)take((size_t)BT * KVD * 2);
  _Float16* Vg  = (_Float16*)take((size_t)BT * KVD * 2);
  _Float16* Qr  = (_Float16*)take((size_t)BT * DIM * 2);
  _Float16* Kr  = (_Float16*)take((size_t)BT * KVD * 2);
  _Float16* Vt  = (_Float16*)take((size_t)BT * KVD * 2);
  _Float16* Og  = (_Float16*)take((size_t)BT * DIM * 2);

  const dim3 blk(256);
  auto grid1 = [](long n) { return dim3((unsigned)((n + 255) / 256)); };
  auto gemm_grid = [](int M, int N) {
    return dim3((unsigned)(((M >> 5) * (N >> 5) + 7) / 8));
  };

  // 1) cast inputs to f16
  cast_f32_f16<<<grid1((long)BT * DIM / 4), blk, 0, stream>>>(x, xh, BT * DIM / 4);
  cast_f32_f16<<<grid1((long)DIM * DIM / 4), blk, 0, stream>>>(wq, wqh, DIM * DIM / 4);
  cast_f32_f16<<<grid1((long)KVD * DIM / 4), blk, 0, stream>>>(wk, wkh, KVD * DIM / 4);
  cast_f32_f16<<<grid1((long)KVD * DIM / 4), blk, 0, stream>>>(wv, wvh, KVD * DIM / 4);
  cast_f32_f16<<<grid1((long)DIM * DIM / 4), blk, 0, stream>>>(wo, woh, DIM * DIM / 4);

  // 2) QKV projections (WMMA)
  gemm_wmma_nt<<<gemm_grid(BT, DIM), blk, 0, stream>>>(xh, wqh, (void*)Qg, BT, DIM, DIM, 1);
  gemm_wmma_nt<<<gemm_grid(BT, KVD), blk, 0, stream>>>(xh, wkh, (void*)Kg, BT, KVD, DIM, 1);
  gemm_wmma_nt<<<gemm_grid(BT, KVD), blk, 0, stream>>>(xh, wvh, (void*)Vg, BT, KVD, DIM, 1);

  // 3) RoPE (+ fold 1/sqrt(64) into Q) and V transpose
  const int totq = BT * NH * HALF_;
  const int totk = BT * NKV * HALF_;
  const int totv = BT * KVD;
  rope_kernel<<<grid1(totq), blk, 0, stream>>>(Qg, cs, sn, Qr, NH, 0.125f, T, totq);
  rope_kernel<<<grid1(totk), blk, 0, stream>>>(Kg, cs, sn, Kr, NKV, 1.0f, T, totk);
  vprep_kernel<<<grid1(totv), blk, 0, stream>>>(Vg, Vt, T, totv);

  // 4) flash attention (WMMA)
  const int attn_waves = 2 * NH * (T >> 4);
  attn_wmma<<<dim3((unsigned)((attn_waves + 7) / 8)), blk, 0, stream>>>(Qr, Kr, Vt, Og, T);

  // 5) output projection -> fp32 d_out (WMMA)
  gemm_wmma_nt<<<gemm_grid(BT, DIM), blk, 0, stream>>>(Og, woh, d_out, BT, DIM, DIM, 0);
}
